// GptOssGroupedExperts_65154653880356
// MI455X (gfx1250) — compile-verified, matrix operand order checked by reading, not run on hardware
//
#include <hip/hip_runtime.h>

// GPT-OSS grouped experts on MI455X (gfx1250):
//   per-expert  H = swiglu(X @ W1^T + b1);  O = H @ W2^T + b2
//   E=8, DIM=HID=2880, T=16384 (2048 tokens/expert)
//
// Fast path (needs ~587 MB ws): one fp32->bf16 conversion pass, then bf16 WMMA
// GEMMs whose tiles are fetched with GLOBAL_LOAD_ASYNC_TO_LDS_B128 (ASYNCcnt),
// double-buffered in LDS. Main loop = v_wmma + ds_load_b128 only.
// Fallback path (ws >= 94 MB): round-1 kernels converting fp32->bf16 in the loader.

#define DIMK 2880
#define HIDK 2880
#define TPE  2048
#define NEXP 8

typedef __bf16 bf16;
typedef __attribute__((ext_vector_type(16))) bf16  v16bf;
typedef __attribute__((ext_vector_type(8)))  bf16  v8bf;
typedef __attribute__((ext_vector_type(4)))  bf16  v4bf;
typedef __attribute__((ext_vector_type(8)))  float v8f;
typedef __attribute__((ext_vector_type(4)))  int   v4i;

#define BM   128      // block M tile
#define BN   96       // block N tile (output columns)
#define BK   64       // K step per LDS stage (2 WMMA k-steps)
#define LDSK 72       // padded LDS row stride in halfs (144 B, 16B-aligned)

// ---------------- CDNA5 async global->LDS copy (16 B per lane) ----------------
#if __has_builtin(__builtin_amdgcn_global_load_async_to_lds_b128)
#define HAS_ASYNC_LDS 1
typedef __attribute__((address_space(1))) v4i g_v4i;   // global int4
typedef __attribute__((address_space(3))) v4i s_v4i;   // LDS int4
__device__ __forceinline__ void cp_async16(const bf16* g, bf16* l) {
  __builtin_amdgcn_global_load_async_to_lds_b128((g_v4i*)g, (s_v4i*)l, 0, 0);
}
#else
#define HAS_ASYNC_LDS 0
__device__ __forceinline__ void cp_async16(const bf16* g, bf16* l) {
  *(v8bf*)l = *(const v8bf*)g;
}
#endif

template <int N> __device__ __forceinline__ void wait_async() {
#if HAS_ASYNC_LDS
#if __has_builtin(__builtin_amdgcn_s_wait_asynccnt)
  __builtin_amdgcn_s_wait_asynccnt(N);
#else
  asm volatile("s_wait_asynccnt %0" ::"i"(N) : "memory");
#endif
#endif
}

// Fragment load from LDS (NT layout). lane&15 -> row, (lane>>4)*8 -> K sub-offset;
// halfs 0..7 = K[ko..ko+7], halfs 8..15 = K[16+ko..16+ko+7].
__device__ __forceinline__ v16bf frag_ld(const bf16* __restrict__ base, int lane,
                                         int rowMul, int kk) {
  const int r  = (lane & 15) * rowMul;
  const int ko = (lane >> 4) << 3;
  const bf16* p = base + r * LDSK + kk + ko;
  v8bf lo = *(const v8bf*)(p);
  v8bf hi = *(const v8bf*)(p + 16);
  return __builtin_shufflevector(lo, hi, 0, 1, 2, 3, 4, 5, 6, 7,
                                         8, 9, 10, 11, 12, 13, 14, 15);
}

// ---------------------------------------------------------------------------
// fp32 -> bf16 conversion pass (grid-stride, float4 granular)
// ---------------------------------------------------------------------------
__global__ __launch_bounds__(256)
void cvt_f32_to_bf16(const float* __restrict__ s, bf16* __restrict__ d, int n4) {
  int i = blockIdx.x * blockDim.x + threadIdx.x;
  int stride = gridDim.x * blockDim.x;
  for (; i < n4; i += stride) {
    float4 v = ((const float4*)s)[i];
    v4bf o = { (bf16)v.x, (bf16)v.y, (bf16)v.z, (bf16)v.w };
    ((v4bf*)d)[i] = o;
  }
}

// ---------------------------------------------------------------------------
// Fast path kernel 1: H = swiglu(Xb @ W1b^T + b1)   (bf16 in, bf16 out)
// Async double-buffered LDS. B slab = W1 rows [2*n0, 2*n0+192) (glu/lin interleaved).
// ---------------------------------------------------------------------------
__global__ __launch_bounds__(256)
void gemm1_swiglu_bf16(const bf16* __restrict__ xb,
                       const bf16* __restrict__ w1b,
                       const float* __restrict__ b1,
                       bf16* __restrict__ hb) {
  __shared__ __align__(16) bf16 As[2][BM * LDSK];
  __shared__ __align__(16) bf16 Bs[2][(2 * BN) * LDSK];

  const int e   = blockIdx.z;
  const int m0  = blockIdx.y * BM;
  const int n0  = blockIdx.x * BN;
  const int tid = threadIdx.x;
  const int lane = tid & 31;
  const int w    = tid >> 5;
  const int wm0  = (w >> 1) * 32;
  const int wn0  = (w & 1) * 48;

  const bf16* __restrict__ xA = xb  + ((size_t)e * TPE + m0) * DIMK;
  const bf16* __restrict__ wB = w1b + ((size_t)e * (2 * HIDK) + 2 * (size_t)n0) * DIMK;

  // 10 async instructions per wave per stage: A=4, B=6
  auto issue = [&](int buf, int k0) {
#pragma unroll
    for (int i = 0; i < 4; ++i) {            // A: 128 rows x 8 chunks
      int c = tid + 256 * i, r = c >> 3, kc = (c & 7) * 8;
      cp_async16(xA + (size_t)r * DIMK + k0 + kc, &As[buf][r * LDSK + kc]);
    }
#pragma unroll
    for (int i = 0; i < 6; ++i) {            // B: 192 rows x 8 chunks
      int c = tid + 256 * i, r = c >> 3, kc = (c & 7) * 8;
      cp_async16(wB + (size_t)r * DIMK + k0 + kc, &Bs[buf][r * LDSK + kc]);
    }
  };

  v8f zero = {};
  v8f accg[2][3], accl[2][3];
#pragma unroll
  for (int i = 0; i < 2; ++i)
#pragma unroll
    for (int j = 0; j < 3; ++j) { accg[i][j] = zero; accl[i][j] = zero; }

  const int nstages = DIMK / BK;   // 45
  issue(0, 0);
  for (int s = 0; s < nstages; ++s) {
    const int cur = s & 1;
    if (s + 1 < nstages) {
      issue(1 - cur, (s + 1) * BK);
      wait_async<10>();            // in-order: <=10 outstanding => stage s landed
    } else {
      wait_async<0>();
    }
    __syncthreads();

#pragma unroll
    for (int kk = 0; kk < BK; kk += 32) {
      v16bf af[2];
#pragma unroll
      for (int mi = 0; mi < 2; ++mi)
        af[mi] = frag_ld(&As[cur][(wm0 + mi * 16) * LDSK], lane, 1, kk);
#pragma unroll
      for (int ni = 0; ni < 3; ++ni) {
        const bf16* bbase = &Bs[cur][(2 * (wn0 + ni * 16)) * LDSK];
        v16bf bg = frag_ld(bbase,        lane, 2, kk);   // even rows: glu
        v16bf bl = frag_ld(bbase + LDSK, lane, 2, kk);   // odd rows:  lin
#pragma unroll
        for (int mi = 0; mi < 2; ++mi) {
          accg[mi][ni] = __builtin_amdgcn_wmma_f32_16x16x32_bf16(
              false, af[mi], false, bg, (short)0, accg[mi][ni], false, false);
          accl[mi][ni] = __builtin_amdgcn_wmma_f32_16x16x32_bf16(
              false, af[mi], false, bl, (short)0, accl[mi][ni], false, false);
        }
      }
    }
    __syncthreads();   // all waves done reading buf[cur] before it is refilled
  }

  const float* __restrict__ bias = b1 + (size_t)e * (2 * HIDK);
  const int rbase = (lane >> 4) << 3;
#pragma unroll
  for (int ni = 0; ni < 3; ++ni) {
    int j = n0 + wn0 + ni * 16 + (lane & 15);
    float bg = bias[2 * j];
    float bl = bias[2 * j + 1];
#pragma unroll
    for (int mi = 0; mi < 2; ++mi) {
#pragma unroll
      for (int r = 0; r < 8; ++r) {
        int m = m0 + wm0 + mi * 16 + r + rbase;
        float g = accg[mi][ni][r] + bg;
        float l = accl[mi][ni][r] + bl;
        g = fminf(g, 7.0f);
        l = fminf(fmaxf(l, -7.0f), 7.0f);
        float act = g * (1.0f / (1.0f + __expf(-1.702f * g)));
        hb[((size_t)e * TPE + m) * HIDK + j] = (bf16)(act * (l + 1.0f));
      }
    }
  }
}

// ---------------------------------------------------------------------------
// Fast path kernel 2: O = H @ W2b^T + b2   (bf16 in, fp32 out)
// ---------------------------------------------------------------------------
__global__ __launch_bounds__(256)
void gemm2_bf16(const bf16* __restrict__ hb,
                const bf16* __restrict__ w2b,
                const float* __restrict__ b2,
                float* __restrict__ out) {
  __shared__ __align__(16) bf16 As[2][BM * LDSK];
  __shared__ __align__(16) bf16 Bs[2][BN * LDSK];

  const int e   = blockIdx.z;
  const int m0  = blockIdx.y * BM;
  const int n0  = blockIdx.x * BN;
  const int tid = threadIdx.x;
  const int lane = tid & 31;
  const int w    = tid >> 5;
  const int wm0  = (w >> 1) * 32;
  const int wn0  = (w & 1) * 48;

  const bf16* __restrict__ hA = hb  + ((size_t)e * TPE + m0) * HIDK;
  const bf16* __restrict__ wB = w2b + ((size_t)e * DIMK + n0) * HIDK;

  // 7 async instructions per wave per stage: A=4, B=3
  auto issue = [&](int buf, int k0) {
#pragma unroll
    for (int i = 0; i < 4; ++i) {
      int c = tid + 256 * i, r = c >> 3, kc = (c & 7) * 8;
      cp_async16(hA + (size_t)r * HIDK + k0 + kc, &As[buf][r * LDSK + kc]);
    }
#pragma unroll
    for (int i = 0; i < 3; ++i) {            // B: 96 rows x 8 chunks
      int c = tid + 256 * i, r = c >> 3, kc = (c & 7) * 8;
      cp_async16(wB + (size_t)r * HIDK + k0 + kc, &Bs[buf][r * LDSK + kc]);
    }
  };

  v8f zero = {};
  v8f acc[2][3];
#pragma unroll
  for (int i = 0; i < 2; ++i)
#pragma unroll
    for (int j = 0; j < 3; ++j) acc[i][j] = zero;

  const int nstages = HIDK / BK;
  issue(0, 0);
  for (int s = 0; s < nstages; ++s) {
    const int cur = s & 1;
    if (s + 1 < nstages) {
      issue(1 - cur, (s + 1) * BK);
      wait_async<7>();
    } else {
      wait_async<0>();
    }
    __syncthreads();

#pragma unroll
    for (int kk = 0; kk < BK; kk += 32) {
      v16bf af[2];
#pragma unroll
      for (int mi = 0; mi < 2; ++mi)
        af[mi] = frag_ld(&As[cur][(wm0 + mi * 16) * LDSK], lane, 1, kk);
#pragma unroll
      for (int ni = 0; ni < 3; ++ni) {
        v16bf bfr = frag_ld(&Bs[cur][(wn0 + ni * 16) * LDSK], lane, 1, kk);
#pragma unroll
        for (int mi = 0; mi < 2; ++mi)
          acc[mi][ni] = __builtin_amdgcn_wmma_f32_16x16x32_bf16(
              false, af[mi], false, bfr, (short)0, acc[mi][ni], false, false);
      }
    }
    __syncthreads();
  }

  const float* __restrict__ bias = b2 + (size_t)e * DIMK;
  const int rbase = (lane >> 4) << 3;
#pragma unroll
  for (int ni = 0; ni < 3; ++ni) {
    int j = n0 + wn0 + ni * 16 + (lane & 15);
    float bb = bias[j];
#pragma unroll
    for (int mi = 0; mi < 2; ++mi) {
#pragma unroll
      for (int r = 0; r < 8; ++r) {
        int m = m0 + wm0 + mi * 16 + r + rbase;
        out[((size_t)e * TPE + m) * DIMK + j] = acc[mi][ni][r] + bb;
      }
    }
  }
}

// ---------------------------------------------------------------------------
// Fallback kernels (small ws): fp32 operands, convert in the LDS loader.
// ---------------------------------------------------------------------------
__global__ __launch_bounds__(256)
void gpt_oss_mlp1_swiglu(const float* __restrict__ x,
                         const float* __restrict__ w1,
                         const float* __restrict__ b1,
                         bf16* __restrict__ hb) {
  __shared__ __align__(16) bf16 As[BM * LDSK];
  __shared__ __align__(16) bf16 Bs[(2 * BN) * LDSK];

  const int e   = blockIdx.z;
  const int m0  = blockIdx.y * BM;
  const int n0  = blockIdx.x * BN;
  const int tid = threadIdx.x;
  const int lane = tid & 31;
  const int w    = tid >> 5;
  const int wm0  = (w >> 1) * 32;
  const int wn0  = (w & 1) * 48;

  const float* __restrict__ xA = x  + ((size_t)e * TPE + m0) * DIMK;
  const float* __restrict__ wB = w1 + ((size_t)e * (2 * HIDK) + 2 * (size_t)n0) * DIMK;

  v8f zero = {};
  v8f accg[2][3], accl[2][3];
#pragma unroll
  for (int i = 0; i < 2; ++i)
#pragma unroll
    for (int j = 0; j < 3; ++j) { accg[i][j] = zero; accl[i][j] = zero; }

  for (int k0 = 0; k0 < DIMK; k0 += BK) {
#pragma unroll
    for (int i = 0; i < 8; ++i) {
      int c = tid + 256 * i, r = c >> 4, kc = (c & 15) * 4;
      float4 v = *(const float4*)(xA + (size_t)r * DIMK + k0 + kc);
      v4bf o = { (bf16)v.x, (bf16)v.y, (bf16)v.z, (bf16)v.w };
      *(v4bf*)(As + r * LDSK + kc) = o;
    }
#pragma unroll
    for (int i = 0; i < 12; ++i) {
      int c = tid + 256 * i, r = c >> 4, kc = (c & 15) * 4;
      float4 v = *(const float4*)(wB + (size_t)r * DIMK + k0 + kc);
      v4bf o = { (bf16)v.x, (bf16)v.y, (bf16)v.z, (bf16)v.w };
      *(v4bf*)(Bs + r * LDSK + kc) = o;
    }
    __syncthreads();

#pragma unroll
    for (int kk = 0; kk < BK; kk += 32) {
      v16bf af[2];
#pragma unroll
      for (int mi = 0; mi < 2; ++mi)
        af[mi] = frag_ld(As + (wm0 + mi * 16) * LDSK, lane, 1, kk);
#pragma unroll
      for (int ni = 0; ni < 3; ++ni) {
        const bf16* bbase = Bs + (2 * (wn0 + ni * 16)) * LDSK;
        v16bf bg = frag_ld(bbase,        lane, 2, kk);
        v16bf bl = frag_ld(bbase + LDSK, lane, 2, kk);
#pragma unroll
        for (int mi = 0; mi < 2; ++mi) {
          accg[mi][ni] = __builtin_amdgcn_wmma_f32_16x16x32_bf16(
              false, af[mi], false, bg, (short)0, accg[mi][ni], false, false);
          accl[mi][ni] = __builtin_amdgcn_wmma_f32_16x16x32_bf16(
              false, af[mi], false, bl, (short)0, accl[mi][ni], false, false);
        }
      }
    }
    __syncthreads();
  }

  const float* __restrict__ bias = b1 + (size_t)e * (2 * HIDK);
  const int rbase = (lane >> 4) << 3;
#pragma unroll
  for (int ni = 0; ni < 3; ++ni) {
    int j = n0 + wn0 + ni * 16 + (lane & 15);
    float bg = bias[2 * j];
    float bl = bias[2 * j + 1];
#pragma unroll
    for (int mi = 0; mi < 2; ++mi) {
#pragma unroll
      for (int r = 0; r < 8; ++r) {
        int m = m0 + wm0 + mi * 16 + r + rbase;
        float g = accg[mi][ni][r] + bg;
        float l = accl[mi][ni][r] + bl;
        g = fminf(g, 7.0f);
        l = fminf(fmaxf(l, -7.0f), 7.0f);
        float act = g * (1.0f / (1.0f + __expf(-1.702f * g)));
        hb[((size_t)e * TPE + m) * HIDK + j] = (bf16)(act * (l + 1.0f));
      }
    }
  }
}

__global__ __launch_bounds__(256)
void gpt_oss_mlp2(const bf16* __restrict__ hb,
                  const float* __restrict__ w2,
                  const float* __restrict__ b2,
                  float* __restrict__ out) {
  __shared__ __align__(16) bf16 As[BM * LDSK];
  __shared__ __align__(16) bf16 Bs[BN * LDSK];

  const int e   = blockIdx.z;
  const int m0  = blockIdx.y * BM;
  const int n0  = blockIdx.x * BN;
  const int tid = threadIdx.x;
  const int lane = tid & 31;
  const int w    = tid >> 5;
  const int wm0  = (w >> 1) * 32;
  const int wn0  = (w & 1) * 48;

  const bf16*  __restrict__ hA = hb + ((size_t)e * TPE + m0) * HIDK;
  const float* __restrict__ wB = w2 + ((size_t)e * DIMK + n0) * HIDK;

  v8f zero = {};
  v8f acc[2][3];
#pragma unroll
  for (int i = 0; i < 2; ++i)
#pragma unroll
    for (int j = 0; j < 3; ++j) acc[i][j] = zero;

  for (int k0 = 0; k0 < HIDK; k0 += BK) {
#pragma unroll
    for (int i = 0; i < 4; ++i) {
      int c = tid + 256 * i, r = c >> 3, kc = (c & 7) * 8;
      v8bf v = *(const v8bf*)(hA + (size_t)r * HIDK + k0 + kc);
      *(v8bf*)(As + r * LDSK + kc) = v;
    }
#pragma unroll
    for (int i = 0; i < 6; ++i) {
      int c = tid + 256 * i, r = c >> 4, kc = (c & 15) * 4;
      float4 v = *(const float4*)(wB + (size_t)r * HIDK + k0 + kc);
      v4bf o = { (bf16)v.x, (bf16)v.y, (bf16)v.z, (bf16)v.w };
      *(v4bf*)(Bs + r * LDSK + kc) = o;
    }
    __syncthreads();

#pragma unroll
    for (int kk = 0; kk < BK; kk += 32) {
      v16bf af[2];
#pragma unroll
      for (int mi = 0; mi < 2; ++mi)
        af[mi] = frag_ld(As + (wm0 + mi * 16) * LDSK, lane, 1, kk);
#pragma unroll
      for (int ni = 0; ni < 3; ++ni) {
        v16bf bfr = frag_ld(Bs + (wn0 + ni * 16) * LDSK, lane, 1, kk);
#pragma unroll
        for (int mi = 0; mi < 2; ++mi)
          acc[mi][ni] = __builtin_amdgcn_wmma_f32_16x16x32_bf16(
              false, af[mi], false, bfr, (short)0, acc[mi][ni], false, false);
      }
    }
    __syncthreads();
  }

  const float* __restrict__ bias = b2 + (size_t)e * DIMK;
  const int rbase = (lane >> 4) << 3;
#pragma unroll
  for (int ni = 0; ni < 3; ++ni) {
    int j = n0 + wn0 + ni * 16 + (lane & 15);
    float bb = bias[j];
#pragma unroll
    for (int mi = 0; mi < 2; ++mi) {
#pragma unroll
      for (int r = 0; r < 8; ++r) {
        int m = m0 + wm0 + mi * 16 + r + rbase;
        out[((size_t)e * TPE + m) * DIMK + j] = acc[mi][ni][r] + bb;
      }
    }
  }
}

extern "C" void kernel_launch(void* const* d_in, const int* in_sizes, int n_in,
                              void* d_out, int out_size, void* d_ws, size_t ws_size,
                              hipStream_t stream) {
  const float* x  = (const float*)d_in[0];
  // d_in[1] = num_tokens_per_expert (constant T/E here; unused)
  const float* w1 = (const float*)d_in[2];
  const float* b1 = (const float*)d_in[3];
  const float* w2 = (const float*)d_in[4];
  const float* b2 = (const float*)d_in[5];
  float* out = (float*)d_out;

  const size_t HB  = (size_t)NEXP * TPE * HIDK * sizeof(bf16);        // 94.4 MB
  const size_t XB  = (size_t)NEXP * TPE * DIMK * sizeof(bf16);        // 94.4 MB
  const size_t W1B = (size_t)NEXP * 2 * HIDK * DIMK * sizeof(bf16);   // 265.4 MB
  const size_t W2B = (size_t)NEXP * DIMK * HIDK * sizeof(bf16);       // 132.7 MB

  bf16* hb = (bf16*)d_ws;
  dim3 blk(256);
  dim3 grid1(HIDK / BN, TPE / BM, NEXP);  // 30 x 16 x 8
  dim3 grid2(DIMK / BN, TPE / BM, NEXP);

  if (ws_size >= HB + XB + W1B + W2B) {
    char* p = (char*)d_ws;
    bf16* xb  = (bf16*)(p + HB);
    bf16* w1b = (bf16*)(p + HB + XB);
    bf16* w2b = (bf16*)(p + HB + XB + W1B);

    cvt_f32_to_bf16<<<4096, blk, 0, stream>>>(x,  xb,  (int)(XB  / sizeof(bf16) / 4));
    cvt_f32_to_bf16<<<4096, blk, 0, stream>>>(w1, w1b, (int)(W1B / sizeof(bf16) / 4));
    cvt_f32_to_bf16<<<4096, blk, 0, stream>>>(w2, w2b, (int)(W2B / sizeof(bf16) / 4));

    gemm1_swiglu_bf16<<<grid1, blk, 0, stream>>>(xb, w1b, b1, hb);
    gemm2_bf16<<<grid2, blk, 0, stream>>>(hb, w2b, b2, out);
  } else {
    gpt_oss_mlp1_swiglu<<<grid1, blk, 0, stream>>>(x, w1, b1, hb);
    gpt_oss_mlp2<<<grid2, blk, 0, stream>>>(hb, w2, b2, out);
  }
}
